// GINNNet_34067680592556
// MI455X (gfx1250) — compile-verified, matrix operand order checked by reading, not run on hardware
//
#include <hip/hip_runtime.h>

typedef __attribute__((ext_vector_type(16))) _Float16 v16h;
typedef __attribute__((ext_vector_type(8)))  _Float16 v8h;
typedef __attribute__((ext_vector_type(8)))  float    v8f;

#define IN_DIM   128
#define EDGE_DIM 64
#define CAT_DIM  192   // IN_DIM + EDGE_DIM
#define HID_DIM  256
#define OUT_DIM  128
#define N_EDGES  640000
#define N_NODES  10000

// LDS row strides (in halves), padded to spread banks (row stride multiple of 16B,
// not a multiple of 64 dwords)
#define MH_STRIDE 200
#define HH_STRIDE 264

// Load one 16-bit A/B fragment group for v_wmma_*_16x16x32_*:
// p0 points at the 8 contiguous halves for this lane's low-K group;
// the high-K group (K+16) is 16 halves further.
__device__ __forceinline__ v16h frag_ld(const _Float16* p0) {
  v8h lo = *(const v8h*)(p0);
  v8h hi = *(const v8h*)(p0 + 16);
  v16h r;
#pragma unroll
  for (int i = 0; i < 8; ++i) { r[i] = lo[i]; r[i + 8] = hi[i]; }
  return r;
}

__global__ void __launch_bounds__(256) gine_zero_out(float* __restrict__ out) {
  int i = blockIdx.x * blockDim.x + threadIdx.x;   // 320000 threads, 4 floats each
  float4 z; z.x = 0.f; z.y = 0.f; z.z = 0.f; z.w = 0.f;
  ((float4*)out)[i] = z;
}

__global__ void __launch_bounds__(256) gine_cvt_weights(
    const float* __restrict__ W1, const float* __restrict__ W2,
    _Float16* __restrict__ W1h, _Float16* __restrict__ W2h) {
  int i = blockIdx.x * blockDim.x + threadIdx.x;   // 81920 = 320*256 exactly
  if (i < HID_DIM * CAT_DIM) {
    W1h[i] = (_Float16)W1[i];
  } else {
    int j = i - HID_DIM * CAT_DIM;                 // j < OUT_DIM*HID_DIM
    W2h[j] = (_Float16)W2[j];
  }
}

// One block (8 waves, wave32) per 16-edge tile.
__global__ void __launch_bounds__(256) gine_tile_kernel(
    const float* __restrict__ x, const long long* __restrict__ ei,
    const float* __restrict__ ea,
    const _Float16* __restrict__ W1h, const float* __restrict__ b1,
    const _Float16* __restrict__ W2h, const float* __restrict__ b2,
    float* __restrict__ out) {
  __shared__ __align__(16) _Float16 Mh[16 * MH_STRIDE];  // concat message tile, f16
  __shared__ __align__(16) _Float16 Hh[16 * HH_STRIDE];  // hidden tile, f16
  __shared__ int srcLds[16];
  __shared__ int dstLds[16];

  const int tid   = threadIdx.x;
  const int lane  = tid & 31;
  const int wave  = tid >> 5;
  const int ebase = blockIdx.x * 16;

  if (tid < 16) {
    srcLds[tid] = (int)ei[ebase + tid];            // edge_index[0]
    dstLds[tid] = (int)ei[N_EDGES + ebase + tid];  // edge_index[1]
  }
  __syncthreads();

  // ---- Stage M = [x[src] | edge_attr] as f16 into LDS; wave handles 2 rows ----
#pragma unroll
  for (int rr = 0; rr < 2; ++rr) {
    const int r = wave * 2 + rr;
    const int s = srcLds[r];
    _Float16* mrow = &Mh[r * MH_STRIDE];
    const float4 xv = *(const float4*)(x + (size_t)s * IN_DIM + lane * 4);
    mrow[lane * 4 + 0] = (_Float16)xv.x;
    mrow[lane * 4 + 1] = (_Float16)xv.y;
    mrow[lane * 4 + 2] = (_Float16)xv.z;
    mrow[lane * 4 + 3] = (_Float16)xv.w;
    if (lane < 16) {
      const float4 av = *(const float4*)(ea + (size_t)(ebase + r) * EDGE_DIM + lane * 4);
      mrow[IN_DIM + lane * 4 + 0] = (_Float16)av.x;
      mrow[IN_DIM + lane * 4 + 1] = (_Float16)av.y;
      mrow[IN_DIM + lane * 4 + 2] = (_Float16)av.z;
      mrow[IN_DIM + lane * 4 + 3] = (_Float16)av.w;
    }
  }
  __syncthreads();

  const int n     = lane & 15;                 // A row (M) for A-frags / B column (N)
  const int khalf = (lane & 16) ? 8 : 0;       // per-lane K sub-group offset
  const int mbase = (lane & 16) ? 8 : 0;       // C/D: vgpr v holds row v (+8 for hi lanes)

  // ---- Layer 1: H = relu(M @ W1^T + b1). Wave computes col tiles j0, j0+8 ----
  {
    const int j0 = wave;
    const int j1 = wave + 8;
    v8f acc0 = {}; v8f acc1 = {};
#pragma unroll
    for (int kk = 0; kk < CAT_DIM / 32; ++kk) {
      const int k0 = kk * 32;
      v16h a  = frag_ld(&Mh[n * MH_STRIDE + k0 + khalf]);
      v16h bA = frag_ld(W1h + (size_t)(j0 * 16 + n) * CAT_DIM + k0 + khalf);
      v16h bB = frag_ld(W1h + (size_t)(j1 * 16 + n) * CAT_DIM + k0 + khalf);
      acc0 = __builtin_amdgcn_wmma_f32_16x16x32_f16(false, a, false, bA,
                                                    (short)0, acc0, false, false);
      acc1 = __builtin_amdgcn_wmma_f32_16x16x32_f16(false, a, false, bB,
                                                    (short)0, acc1, false, false);
    }
    const float bb0 = b1[j0 * 16 + n];
    const float bb1 = b1[j1 * 16 + n];
#pragma unroll
    for (int v = 0; v < 8; ++v) {
      const int m = mbase + v;
      Hh[m * HH_STRIDE + j0 * 16 + n] = (_Float16)fmaxf(acc0[v] + bb0, 0.0f);
      Hh[m * HH_STRIDE + j1 * 16 + n] = (_Float16)fmaxf(acc1[v] + bb1, 0.0f);
    }
  }
  __syncthreads();

  // ---- Layer 2: msg = H @ W2^T + b2. Wave computes out col tile `wave` ----
  {
    const int j2 = wave;
    v8f acc = {};
#pragma unroll
    for (int kk = 0; kk < HID_DIM / 32; ++kk) {
      const int k0 = kk * 32;
      v16h a = frag_ld(&Hh[n * HH_STRIDE + k0 + khalf]);
      v16h b = frag_ld(W2h + (size_t)(j2 * 16 + n) * HID_DIM + k0 + khalf);
      acc = __builtin_amdgcn_wmma_f32_16x16x32_f16(false, a, false, b,
                                                   (short)0, acc, false, false);
    }
    const float bb2 = b2[j2 * 16 + n];
#pragma unroll
    for (int v = 0; v < 8; ++v) {
      const int m    = mbase + v;
      const int node = dstLds[m];
      atomicAdd(out + (size_t)node * OUT_DIM + j2 * 16 + n, acc[v] + bb2);
    }
  }
}

extern "C" void kernel_launch(void* const* d_in, const int* in_sizes, int n_in,
                              void* d_out, int out_size, void* d_ws, size_t ws_size,
                              hipStream_t stream) {
  const float*     x  = (const float*)d_in[0];
  const long long* ei = (const long long*)d_in[1];   // int64 edge_index [2, E]
  const float*     ea = (const float*)d_in[2];
  const float*     W1 = (const float*)d_in[3];
  const float*     b1 = (const float*)d_in[4];
  const float*     W2 = (const float*)d_in[5];
  const float*     b2 = (const float*)d_in[6];
  float*           out = (float*)d_out;

  _Float16* W1h = (_Float16*)d_ws;                                    // 98304 B
  _Float16* W2h = (_Float16*)((char*)d_ws +
                              (size_t)HID_DIM * CAT_DIM * sizeof(_Float16)); // +65536 B

  // out accumulates via atomics -> must start at zero every call.
  gine_zero_out<<<(N_NODES * OUT_DIM / 4) / 256, 256, 0, stream>>>(out);
  gine_cvt_weights<<<(HID_DIM * CAT_DIM + OUT_DIM * HID_DIM) / 256, 256, 0, stream>>>(
      W1, W2, W1h, W2h);
  gine_tile_kernel<<<N_EDGES / 16, 256, 0, stream>>>(x, ei, ea, W1h, b1, W2h, b2, out);
}
